// SelfAttention_52527450030549
// MI455X (gfx1250) — matrix operand order repaired and checked by reading, not compiled, not run on hardware
//
#include <hip/hip_runtime.h>
#include <hip/hip_bf16.h>

// ---------------------------------------------------------------------------
// T5-style self-attention for MI455X (gfx1250, wave32, WMMA bf16 path).
//   out = ( softmax( (xWq^T)(xWk^T)^T + bias ) (xWv^T) ) Wo^T
// - All GEMMs: v_wmma_f32_16x16x32_bf16 (bf16 in, fp32 accumulate).
// - Attention: flash-style online softmax; bias fused (prefetched); scores
//   never touch HBM. V is produced pre-transposed per head so the P*V
//   B-operand is a contiguous 2x b128 load (same shape as the K operand).
// - P restage (C-layout f32 -> A-layout bf16) goes through wave-private LDS
//   with wave-level ordering only (LDS is in-order per wave on CDNA5).
// ---------------------------------------------------------------------------

typedef __bf16 bf16_t;
typedef __attribute__((ext_vector_type(16))) __bf16 v16bf;
typedef __attribute__((ext_vector_type(8)))  __bf16 v8bf;
typedef __attribute__((ext_vector_type(8)))  float  v8f;

#define D_MODEL 1024
#define N_HEADS 16
#define DK      64
#define BATCH   2
#define SEQ     2048
#define MROWS   (BATCH * SEQ)   // 4096

#define WMMA_BF16(A, B, C) \
  __builtin_amdgcn_wmma_f32_16x16x32_bf16(false, (A), false, (B), (short)0, (C), false, false)

// A-operand fragment (16x32, M x K): lane L holds row M = L&15.
// element e -> K = 16*(L>>4) + e
// caller passes p = rowPtr + k0 + 16*(lane>>4); loads 16 contiguous bf16.
__device__ __forceinline__ v16bf load_a_frag(const bf16_t* p) {
  v8bf lo = *(const v8bf*)p;
  v8bf hi = *(const v8bf*)(p + 8);
  v16bf r;
#pragma unroll
  for (int i = 0; i < 8; ++i) { r[i] = lo[i]; r[i + 8] = hi[i]; }
  return r;
}

// B-operand fragment (32x16, K x N): lane L holds column N = L&15.
// element e -> K = 16*(L>>4) + e (rows of B striped across lanes per VGPR).
// caller passes p = colPtr + k0 + 16*(lane>>4); loads 16 contiguous bf16.
__device__ __forceinline__ v16bf load_b_frag(const bf16_t* p) {
  v8bf lo = *(const v8bf*)p;
  v8bf hi = *(const v8bf*)(p + 8);
  v16bf r;
#pragma unroll
  for (int i = 0; i < 8; ++i) { r[i] = lo[i]; r[i + 8] = hi[i]; }
  return r;
}

// Wave-private LDS exchange ordering: lanes of a wave32 run in lockstep and
// LDS ops from one wave complete in order (DScnt), so a compiler scheduling
// barrier + memory clobber is sufficient (no s_barrier needed).
__device__ __forceinline__ void wave_sync() {
  asm volatile("" ::: "memory");
  __builtin_amdgcn_wave_barrier();
  asm volatile("" ::: "memory");
}

// ---------------------------------------------------------------------------
__global__ void cast_f32_bf16(const float* __restrict__ src,
                              bf16_t* __restrict__ dst, int n) {
  int i = blockIdx.x * blockDim.x + threadIdx.x;
  if (i < n) dst[i] = (bf16_t)src[i];
}

// ---------------------------------------------------------------------------
// C[M,N] = A[M,K] * Bw[N,K]^T  (y = x @ W^T). Block = 8 waves; each wave does
// a 16(M) x 64(N) strip: one A fragment reused by 4 B fragments per k-step.
// Store modes: Cf (fp32 row-major) | Cb (bf16 row-major) | CbT (bf16,
// per-head transposed: Vt[((b*16+h)*64+d)*SEQ + seq], for the V projection).
__global__ __launch_bounds__(256) void gemm_xwT(const bf16_t* __restrict__ A,
                                                const bf16_t* __restrict__ Bw,
                                                bf16_t* __restrict__ Cb,
                                                float* __restrict__ Cf,
                                                bf16_t* __restrict__ CbT,
                                                int M, int N, int K) {
  const int lane = threadIdx.x & 31;
  const int wv   = threadIdx.x >> 5;
  const int nn   = lane & 15;
  const int half = lane >> 4;
  const int m0   = blockIdx.y * 16;
  const int n0   = blockIdx.x * 512 + wv * 64;

  const bf16_t* arow = A + (size_t)(m0 + nn) * K + 16 * half;
  const bf16_t* bcol[4];
#pragma unroll
  for (int f = 0; f < 4; ++f)
    bcol[f] = Bw + (size_t)(n0 + f * 16 + nn) * K + 16 * half;

  v8f acc[4] = {v8f{}, v8f{}, v8f{}, v8f{}};
  for (int k0 = 0; k0 < K; k0 += 32) {
    v16bf a = load_a_frag(arow + k0);
#pragma unroll
    for (int f = 0; f < 4; ++f)
      acc[f] = WMMA_BF16(a, load_b_frag(bcol[f] + k0), acc[f]);
  }

  // C/D layout: VGPR r, lane L -> M = m0 + r + 8*(L>>4), N = n0 + f*16 + (L&15)
  const int mb = 8 * half;
#pragma unroll
  for (int r = 0; r < 8; ++r) {
    const int m = m0 + mb + r;
    if (Cf) {
      size_t row = (size_t)m * N + n0 + nn;
#pragma unroll
      for (int f = 0; f < 4; ++f) Cf[row + f * 16] = acc[f][r];
    } else if (Cb) {
      size_t row = (size_t)m * N + n0 + nn;
#pragma unroll
      for (int f = 0; f < 4; ++f) Cb[row + f * 16] = (bf16_t)acc[f][r];
    } else {
      const int bidx = m >> 11;           // m / SEQ
      const int seq  = m & (SEQ - 1);
#pragma unroll
      for (int f = 0; f < 4; ++f) {
        const int col = n0 + f * 16 + nn;
        const int hh = col >> 6, d = col & (DK - 1);
        CbT[((size_t)(bidx * N_HEADS + hh) * DK + d) * SEQ + seq] = (bf16_t)acc[f][r];
      }
    }
  }
}

// ---------------------------------------------------------------------------
// Flash attention per (b,h): each wave owns a 16-query tile, streams keys in
// 32-wide steps. S = Q*K^T (4 wmma) + bias, online softmax, P restaged via
// wave-private LDS into A-layout, ctx += P*V (4 wmma, V read from per-head
// transposed Vt so B fragments are contiguous). ctx written bf16 [B*SEQ,D].
__global__ __launch_bounds__(128) void attn_flash(const bf16_t* __restrict__ Qb,
                                                  const bf16_t* __restrict__ Kb,
                                                  const bf16_t* __restrict__ Vt,
                                                  const float* __restrict__ bias,
                                                  bf16_t* __restrict__ Ctx) {
  __shared__ __align__(16) bf16_t pbuf[4][16 * 32];  // per-wave P staging (1KB)

  const int lane = threadIdx.x & 31;
  const int wv   = threadIdx.x >> 5;
  const int nn   = lane & 15;
  const int half = lane >> 4;
  const int b    = blockIdx.y >> 4;
  const int h    = blockIdx.y & 15;
  const int q0   = blockIdx.x * 64 + wv * 16;
  const size_t rowbase = (size_t)b * SEQ;

  // Q A-fragments for the two 32-wide dk chunks (loaded once).
  const bf16_t* qrow = Qb + (rowbase + q0 + nn) * D_MODEL + h * DK + 16 * half;
  v16bf qa0 = load_a_frag(qrow + 0);
  v16bf qa1 = load_a_frag(qrow + 32);

  const float*  biasrow = bias + (size_t)h * SEQ * SEQ;
  const bf16_t* vthead  = Vt + (size_t)(b * N_HEADS + h) * DK * SEQ;  // [64][SEQ]

  float rmax[8], rsum[8];
#pragma unroll
  for (int r = 0; r < 8; ++r) { rmax[r] = -3.0e38f; rsum[r] = 0.0f; }
  v8f o[4] = {v8f{}, v8f{}, v8f{}, v8f{}};

  const int mb = 8 * half;
  for (int k0 = 0; k0 < SEQ; k0 += 32) {
    // ---- S tile (16 queries x 32 keys), fp32 accumulate ----
    v8f s0 = {}, s1 = {};
    {
      const bf16_t* kr0 = Kb + (rowbase + k0 + nn) * D_MODEL + h * DK + 16 * half;
      const bf16_t* kr1 = Kb + (rowbase + k0 + 16 + nn) * D_MODEL + h * DK + 16 * half;
      s0 = WMMA_BF16(qa0, load_b_frag(kr0 + 0),  s0);
      s0 = WMMA_BF16(qa1, load_b_frag(kr0 + 32), s0);
      s1 = WMMA_BF16(qa0, load_b_frag(kr1 + 0),  s1);
      s1 = WMMA_BF16(qa1, load_b_frag(kr1 + 32), s1);
    }
    // ---- bias add (+ prefetch bias lines 64 keys (two tiles) ahead) ----
#pragma unroll
    for (int r = 0; r < 8; ++r) {
      size_t bi = (size_t)(q0 + mb + r) * SEQ + k0 + nn;
      s0[r] += biasrow[bi];
      s1[r] += biasrow[bi + 16];
      if (k0 + 64 < SEQ) __builtin_prefetch(biasrow + bi + 64, 0, 3);
    }
    // ---- online softmax (rows live in 16-lane halves of the C layout) ----
#pragma unroll
    for (int r = 0; r < 8; ++r) {
      float v = fmaxf(s0[r], s1[r]);
      v = fmaxf(v, __shfl_xor(v, 1, 32));
      v = fmaxf(v, __shfl_xor(v, 2, 32));
      v = fmaxf(v, __shfl_xor(v, 4, 32));
      v = fmaxf(v, __shfl_xor(v, 8, 32));
      float nm   = fmaxf(rmax[r], v);
      float corr = __expf(rmax[r] - nm);
      rmax[r] = nm;
      float p0 = __expf(s0[r] - nm);
      float p1 = __expf(s1[r] - nm);
      s0[r] = p0; s1[r] = p1;
      float ps = p0 + p1;
      ps += __shfl_xor(ps, 1, 32);
      ps += __shfl_xor(ps, 2, 32);
      ps += __shfl_xor(ps, 4, 32);
      ps += __shfl_xor(ps, 8, 32);
      rsum[r] = rsum[r] * corr + ps;
#pragma unroll
      for (int f = 0; f < 4; ++f) o[f][r] *= corr;
    }
    // ---- restage P: C-layout fp32 -> A-layout bf16 via wave-private LDS ----
    wave_sync();  // prior iteration's LDS reads ordered before overwrite
#pragma unroll
    for (int r = 0; r < 8; ++r) {
      pbuf[wv][(mb + r) * 32 + nn]      = (bf16_t)s0[r];
      pbuf[wv][(mb + r) * 32 + 16 + nn] = (bf16_t)s1[r];
    }
    wave_sync();  // writes ordered before reads (in-order LDS, lockstep wave)
    v16bf pa = load_a_frag(&pbuf[wv][nn * 32 + 16 * half]);
    // ---- ctx += P(16x32) * V(32x64): contiguous B fragments from Vt ----
#pragma unroll
    for (int f = 0; f < 4; ++f) {
      const bf16_t* vp = vthead + (size_t)(f * 16 + nn) * SEQ + k0 + 16 * half;
      o[f] = WMMA_BF16(pa, load_b_frag(vp), o[f]);
    }
  }
  // ---- normalize and write ctx (bf16) ----
#pragma unroll
  for (int r = 0; r < 8; ++r) {
    float inv = 1.0f / rsum[r];
    size_t row = (rowbase + q0 + mb + r) * D_MODEL + h * DK + nn;
#pragma unroll
    for (int f = 0; f < 4; ++f)
      Ctx[row + f * 16] = (bf16_t)(o[f][r] * inv);
  }
}

// ---------------------------------------------------------------------------
extern "C" void kernel_launch(void* const* d_in, const int* in_sizes, int n_in,
                              void* d_out, int out_size, void* d_ws, size_t ws_size,
                              hipStream_t stream) {
  const float* x    = (const float*)d_in[0];
  const float* bias = (const float*)d_in[1];
  const float* wq   = (const float*)d_in[2];
  const float* wk   = (const float*)d_in[3];
  const float* wvp  = (const float*)d_in[4];
  const float* wo   = (const float*)d_in[5];
  float* out = (float*)d_out;

  char* ws = (char*)d_ws;  // 48 MB used
  const size_t MB = 1u << 20;
  bf16_t* xb  = (bf16_t*)(ws + 0 * MB);    // 8 MB  [4096,1024]
  bf16_t* wqb = (bf16_t*)(ws + 8 * MB);    // 2 MB each
  bf16_t* wkb = (bf16_t*)(ws + 10 * MB);
  bf16_t* wvb = (bf16_t*)(ws + 12 * MB);
  bf16_t* wob = (bf16_t*)(ws + 14 * MB);
  bf16_t* Qb  = (bf16_t*)(ws + 16 * MB);   // 8 MB each
  bf16_t* Kb  = (bf16_t*)(ws + 24 * MB);
  bf16_t* Vt  = (bf16_t*)(ws + 32 * MB);   // per-head transposed V
  bf16_t* Cx  = (bf16_t*)(ws + 40 * MB);

  const int nx = MROWS * D_MODEL;
  const int nw = D_MODEL * D_MODEL;
  cast_f32_bf16<<<(nx + 255) / 256, 256, 0, stream>>>(x,   xb,  nx);
  cast_f32_bf16<<<(nw + 255) / 256, 256, 0, stream>>>(wq,  wqb, nw);
  cast_f32_bf16<<<(nw + 255) / 256, 256, 0, stream>>>(wk,  wkb, nw);
  cast_f32_bf16<<<(nw + 255) / 256, 256, 0, stream>>>(wvp, wvb, nw);
  cast_f32_bf16<<<(nw + 255) / 256, 256, 0, stream>>>(wo,  wob, nw);

  dim3 gg(D_MODEL / 512, MROWS / 16);  // (2, 256)
  gemm_xwT<<<gg, 256, 0, stream>>>(xb, wqb, Qb,      nullptr, nullptr, MROWS, D_MODEL, D_MODEL);
  gemm_xwT<<<gg, 256, 0, stream>>>(xb, wkb, Kb,      nullptr, nullptr, MROWS, D_MODEL, D_MODEL);
  gemm_xwT<<<gg, 256, 0, stream>>>(xb, wvb, nullptr, nullptr, Vt,      MROWS, D_MODEL, D_MODEL);

  attn_flash<<<dim3(SEQ / 64, BATCH * N_HEADS), 128, 0, stream>>>(Qb, Kb, Vt, bias, Cx);

  gemm_xwT<<<gg, 256, 0, stream>>>(Cx, wob, nullptr, out, nullptr, MROWS, D_MODEL, D_MODEL);
}
